// FastSpeech_43508018709267
// MI455X (gfx1250) — matrix operand order repaired and checked by reading, not compiled, not run on hardware
//
#include <hip/hip_runtime.h>
#include <hip/hip_bf16.h>
#include <math.h>

// ---------------------------------------------------------------------------
// FastSpeech forward for MI455X (gfx1250). All GEMM-like math through
// v_wmma_f32_16x16x32_bf16 (wave32). Double-buffered LDS staging, 128-bit
// global loads, 128-bit LDS fragment loads.
// ---------------------------------------------------------------------------

typedef __attribute__((ext_vector_type(16))) __bf16 v16bf;
typedef __attribute__((ext_vector_type(8)))  __bf16 v8bf;
typedef __attribute__((ext_vector_type(4)))  __bf16 v4bf;
typedef __attribute__((ext_vector_type(8)))  float  v8f;

#define NI_    384
#define NI3_   1152
#define DH_    96
#define NH_    4
#define NB_    4
#define SS_    256
#define TT_    1024
#define NOUT_  80
#define NLAY_  6

__device__ __forceinline__ v8f wmma_bf16(v16bf a, v16bf b, v8f c) {
  return __builtin_amdgcn_wmma_f32_16x16x32_bf16(
      false, a, false, b, (short)0, c, false, false);
}

// Build a 16-element bf16 fragment from two contiguous 8-element (16B) runs.
// ISA A-layout: lane half 'hi' holds K = hi*8..hi*8+7 and 16+hi*8..+7,
// so both runs are contiguous -> two ds_load_b128.
__device__ __forceinline__ v16bf frag2(const __bf16* p0, const __bf16* p1) {
  v8bf lo = *(const v8bf*)p0;
  v8bf hi = *(const v8bf*)p1;
  v16bf r;
#pragma unroll
  for (int j = 0; j < 8; ++j) { r[j] = lo[j]; r[j + 8] = hi[j]; }
  return r;
}

__device__ __forceinline__ v4bf cvt4(float4 v) {
  return v4bf{(__bf16)v.x, (__bf16)v.y, (__bf16)v.z, (__bf16)v.w};
}

// ---------------------------------------------------------------------------
// Block-tiled GEMM: out[N,M] = A[N,K] @ W[K,M] + bias (+ resid).
// Block = 128x64, 8 waves of 32x32 (4 WMMA accumulators each).
// Double-buffered LDS: global loads for slab i+1 issue before computing slab i.
// N multiple of 128, K multiple of 32; M ragged-guarded.
// outT!=0 => store out[(b*M+col)*outT + t], row = b*outT + t.
// ---------------------------------------------------------------------------
__global__ __launch_bounds__(256) void gemm_wmma_kernel(
    const float* __restrict__ A, int N, int K,
    const float* __restrict__ W, int M,
    const float* __restrict__ bias, const float* __restrict__ resid,
    float* __restrict__ out, int outT) {
  __shared__ __align__(16) __bf16 Abf[2][128][32];   // A slabs (rows x k)
  __shared__ __align__(16) __bf16 Bt[2][64][32];     // W slabs transposed (col x k)

  const int tid = threadIdx.x;
  const int lane = tid & 31, wave = tid >> 5;
  const int tmb = (M + 63) >> 6;
  const int n0 = (blockIdx.x / tmb) << 7;
  const int m0 = (blockIdx.x % tmb) << 6;
  const int wr = wave >> 1;             // 0..3 : 32-row subtile
  const int wc = wave & 1;              // 0..1 : 32-col subtile
  const int m = lane & 15, hi = lane >> 4;
  const int nk = K >> 5;

  float4 ar[4];                         // A slab registers (128x32 / 256 thr)
  float4 wr4[2];                        // W slab registers (32x64 / 256 thr)

  auto load_regs = [&](int slab) {
    const int kk = slab << 5;
#pragma unroll
    for (int i = 0; i < 4; ++i) {
      int idx = tid + i * 256;                       // 0..1023
      int row = idx >> 3, kq = (idx & 7) << 2;
      const float* ap = &A[(size_t)(n0 + row) * K + kk + kq];
      ar[i] = *(const float4*)ap;
      if (i == 0 && kk + 32 < K) __builtin_prefetch(ap + 32, 0, 3);
    }
#pragma unroll
    for (int i = 0; i < 2; ++i) {
      int idx = tid + i * 256;                       // 0..511
      int col4 = (idx & 15) << 2, k = idx >> 4;
      int colg = m0 + col4;
      if (colg + 3 < M) {
        wr4[i] = *(const float4*)&W[(size_t)(kk + k) * M + colg];
      } else {
        const float* wp = &W[(size_t)(kk + k) * M];
        wr4[i].x = (colg + 0 < M) ? wp[colg + 0] : 0.0f;
        wr4[i].y = (colg + 1 < M) ? wp[colg + 1] : 0.0f;
        wr4[i].z = (colg + 2 < M) ? wp[colg + 2] : 0.0f;
        wr4[i].w = (colg + 3 < M) ? wp[colg + 3] : 0.0f;
      }
    }
  };
  auto store_lds = [&](int buf) {
#pragma unroll
    for (int i = 0; i < 4; ++i) {
      int idx = tid + i * 256;
      int row = idx >> 3, kq = (idx & 7) << 2;
      *(v4bf*)&Abf[buf][row][kq] = cvt4(ar[i]);
    }
#pragma unroll
    for (int i = 0; i < 2; ++i) {
      int idx = tid + i * 256;
      int col4 = (idx & 15) << 2, k = idx >> 4;
      Bt[buf][col4 + 0][k] = (__bf16)wr4[i].x;
      Bt[buf][col4 + 1][k] = (__bf16)wr4[i].y;
      Bt[buf][col4 + 2][k] = (__bf16)wr4[i].z;
      Bt[buf][col4 + 3][k] = (__bf16)wr4[i].w;
    }
  };

  v8f acc[2][2] = {};                   // [row subtile][col subtile]

  load_regs(0);
  store_lds(0);
  __syncthreads();

  for (int s = 0; s < nk; ++s) {
    const int cur = s & 1;
    if (s + 1 < nk) load_regs(s + 1);   // issue next-slab global loads early

    const __bf16* ar0 = &Abf[cur][wr * 32 + m][0];
    const __bf16* ar1 = &Abf[cur][wr * 32 + 16 + m][0];
    const __bf16* br0 = &Bt[cur][wc * 32 + m][0];
    const __bf16* br1 = &Bt[cur][wc * 32 + 16 + m][0];
    v16bf a0 = frag2(ar0 + hi * 8, ar0 + 16 + hi * 8);
    v16bf a1 = frag2(ar1 + hi * 8, ar1 + 16 + hi * 8);
    v16bf b0 = frag2(br0 + hi * 8, br0 + 16 + hi * 8);
    v16bf b1 = frag2(br1 + hi * 8, br1 + 16 + hi * 8);
    acc[0][0] = wmma_bf16(a0, b0, acc[0][0]);
    acc[0][1] = wmma_bf16(a0, b1, acc[0][1]);
    acc[1][0] = wmma_bf16(a1, b0, acc[1][0]);
    acc[1][1] = wmma_bf16(a1, b1, acc[1][1]);

    if (s + 1 < nk) store_lds(cur ^ 1); // convert + stage into other buffer
    __syncthreads();
  }

  // ---- store (C layout: lane half selects M rows r / 8+r) ----
#pragma unroll
  for (int rs = 0; rs < 2; ++rs) {
#pragma unroll
    for (int cs = 0; cs < 2; ++cs) {
      int col = m0 + wc * 32 + cs * 16 + m;
      if (col >= M) continue;
      float bi = bias ? bias[col] : 0.0f;
#pragma unroll
      for (int r = 0; r < 8; ++r) {
        int row = n0 + wr * 32 + rs * 16 + ((lane < 16) ? r : (8 + r));
        float v = acc[rs][cs][r] + bi;
        if (resid) v += resid[(size_t)row * M + col];
        if (outT) {
          int bidx = row / outT, t = row - bidx * outT;
          out[((size_t)bidx * M + col) * outT + t] = v;
        } else {
          out[(size_t)row * M + col] = v;
        }
      }
    }
  }
}

// ---------------------------------------------------------------------------
// Conv1d (kernel 3, same-pad) + bias + residual. Same 128x64 double-buffered
// tiling; tap x k flattened into one slab pipeline accumulating into the same
// WMMA accumulators. x: [Bn*S, C]; cw: [C_out, C_in, 3]. C multiple of 64.
// ---------------------------------------------------------------------------
__global__ __launch_bounds__(256) void conv_wmma_kernel(
    const float* __restrict__ x, const float* __restrict__ cw,
    const float* __restrict__ cb, float* __restrict__ out,
    int Bn, int S, int C) {
  __shared__ __align__(16) __bf16 Abf[2][128][32];
  __shared__ __align__(16) __bf16 Bt[2][64][32];

  const int tid = threadIdx.x;
  const int lane = tid & 31, wave = tid >> 5;
  const int tmb = C >> 6;
  const int n0 = (blockIdx.x / tmb) << 7;
  const int m0 = (blockIdx.x % tmb) << 6;
  const int wr = wave >> 1, wc = wave & 1;
  const int m = lane & 15, hi = lane >> 4;
  const int nkc = C >> 5;               // k slabs per tap
  const int ns = 3 * nkc;               // total slabs

  float4 ar[4];
  float4 wr4[2];

  auto load_regs = [&](int slab) {
    const int tap = slab / nkc;
    const int kk = (slab - tap * nkc) << 5;
#pragma unroll
    for (int i = 0; i < 4; ++i) {
      int idx = tid + i * 256;
      int row = idx >> 3, kq = (idx & 7) << 2;
      int rowg = n0 + row;
      int b_ = rowg / S, sq = rowg - b_ * S;
      int s2 = sq + tap - 1;
      if ((unsigned)s2 < (unsigned)S) {
        ar[i] = *(const float4*)&x[(size_t)(b_ * S + s2) * C + kk + kq];
      } else {
        ar[i] = float4{0.0f, 0.0f, 0.0f, 0.0f};
      }
    }
#pragma unroll
    for (int i = 0; i < 2; ++i) {
      int idx = tid + i * 256;
      int col4 = (idx & 15) << 2, k = idx >> 4;
      // cw[o][i][tap], o = m0+col, i = kk+k : stride-3 gather
      const float* wp = &cw[((size_t)(m0 + col4) * C + kk + k) * 3 + tap];
      wr4[i].x = wp[0 * C * 3];
      wr4[i].y = wp[1 * C * 3];
      wr4[i].z = wp[2 * C * 3];
      wr4[i].w = wp[3 * C * 3];
    }
  };
  auto store_lds = [&](int buf) {
#pragma unroll
    for (int i = 0; i < 4; ++i) {
      int idx = tid + i * 256;
      int row = idx >> 3, kq = (idx & 7) << 2;
      *(v4bf*)&Abf[buf][row][kq] = cvt4(ar[i]);
    }
#pragma unroll
    for (int i = 0; i < 2; ++i) {
      int idx = tid + i * 256;
      int col4 = (idx & 15) << 2, k = idx >> 4;
      Bt[buf][col4 + 0][k] = (__bf16)wr4[i].x;
      Bt[buf][col4 + 1][k] = (__bf16)wr4[i].y;
      Bt[buf][col4 + 2][k] = (__bf16)wr4[i].z;
      Bt[buf][col4 + 3][k] = (__bf16)wr4[i].w;
    }
  };

  v8f acc[2][2] = {};

  load_regs(0);
  store_lds(0);
  __syncthreads();

  for (int s = 0; s < ns; ++s) {
    const int cur = s & 1;
    if (s + 1 < ns) load_regs(s + 1);

    const __bf16* ar0 = &Abf[cur][wr * 32 + m][0];
    const __bf16* ar1 = &Abf[cur][wr * 32 + 16 + m][0];
    const __bf16* br0 = &Bt[cur][wc * 32 + m][0];
    const __bf16* br1 = &Bt[cur][wc * 32 + 16 + m][0];
    v16bf a0 = frag2(ar0 + hi * 8, ar0 + 16 + hi * 8);
    v16bf a1 = frag2(ar1 + hi * 8, ar1 + 16 + hi * 8);
    v16bf b0 = frag2(br0 + hi * 8, br0 + 16 + hi * 8);
    v16bf b1 = frag2(br1 + hi * 8, br1 + 16 + hi * 8);
    acc[0][0] = wmma_bf16(a0, b0, acc[0][0]);
    acc[0][1] = wmma_bf16(a0, b1, acc[0][1]);
    acc[1][0] = wmma_bf16(a1, b0, acc[1][0]);
    acc[1][1] = wmma_bf16(a1, b1, acc[1][1]);

    if (s + 1 < ns) store_lds(cur ^ 1);
    __syncthreads();
  }

#pragma unroll
  for (int rs = 0; rs < 2; ++rs) {
#pragma unroll
    for (int cs = 0; cs < 2; ++cs) {
      int col = m0 + wc * 32 + cs * 16 + m;
      float bi = cb[col];
#pragma unroll
      for (int r = 0; r < 8; ++r) {
        int row = n0 + wr * 32 + rs * 16 + ((lane < 16) ? r : (8 + r));
        out[(size_t)row * C + col] = acc[rs][cs][r] + bi + x[(size_t)row * C + col];
      }
    }
  }
}

// ---------------------------------------------------------------------------
// Flash-style attention, single wave per (b, head, 16-query tile).
// h: [B*S, 3*NI], per-head layout [q(96) k(96) v(96)]; o: [B*S, NI].
// ---------------------------------------------------------------------------
__global__ __launch_bounds__(32) void attn_wmma_kernel(
    const float* __restrict__ h, float* __restrict__ o, int S) {
  const int lane = threadIdx.x;
  const int q0 = blockIdx.x * 16;
  const int head = blockIdx.y;
  const int b = blockIdx.z;
  const int hoff = head * 3 * DH_;

  __shared__ __align__(16) __bf16 Qs[16][DH_];
  __shared__ __align__(16) __bf16 Ks[16][DH_];
  __shared__ __align__(16) __bf16 VsT[DH_][32];   // V transposed, k 16..31 zero
  __shared__ __align__(16) __bf16 Ps[16][32];     // probs, k 16..31 zero
  __shared__ float  Ss[16][16];
  __shared__ float  Os[16][DH_];
  __shared__ float  mrow[16], lrow[16];

  const float inv_scale = 0.102062072616f;        // 1/sqrt(96)
  const int m = lane & 15, hi = lane >> 4;

  for (int idx = lane; idx < 16 * 24; idx += 32) {
    int r = idx / 24, c4 = (idx - r * 24) * 4;
    float4 q = *(const float4*)&h[(size_t)(b * S + q0 + r) * NI3_ + hoff + c4];
    Qs[r][c4 + 0] = (__bf16)(q.x * inv_scale);
    Qs[r][c4 + 1] = (__bf16)(q.y * inv_scale);
    Qs[r][c4 + 2] = (__bf16)(q.z * inv_scale);
    Qs[r][c4 + 3] = (__bf16)(q.w * inv_scale);
  }
  for (int idx = lane; idx < 16 * DH_; idx += 32) {
    int r = idx / DH_, cc = idx - r * DH_;
    Os[r][cc] = 0.0f;
    VsT[cc][16 + r] = (__bf16)0.0f;               // zero pad k=16..31
  }
  for (int idx = lane; idx < 16 * 16; idx += 32)
    Ps[idx >> 4][16 + (idx & 15)] = (__bf16)0.0f;
  if (lane < 16) { mrow[lane] = -3.0e38f; lrow[lane] = 0.0f; }
  __syncthreads();

  for (int kb = 0; kb < S; kb += 16) {
    for (int idx = lane; idx < 16 * 24; idx += 32) {
      int r = idx / 24, c4 = (idx - r * 24) * 4;
      size_t base = (size_t)(b * S + kb + r) * NI3_ + hoff;
      float4 kv = *(const float4*)&h[base + DH_ + c4];
      float4 vv = *(const float4*)&h[base + 2 * DH_ + c4];
      Ks[r][c4 + 0] = (__bf16)kv.x; Ks[r][c4 + 1] = (__bf16)kv.y;
      Ks[r][c4 + 2] = (__bf16)kv.z; Ks[r][c4 + 3] = (__bf16)kv.w;
      VsT[c4 + 0][r] = (__bf16)vv.x; VsT[c4 + 1][r] = (__bf16)vv.y;
      VsT[c4 + 2][r] = (__bf16)vv.z; VsT[c4 + 3][r] = (__bf16)vv.w;
    }
    __syncthreads();

    // S = Q @ K^T over dh=96 (B(k,n) = K[n][k], contiguous in Ks rows)
    v8f c = {};
    for (int kk = 0; kk < DH_; kk += 32) {
      v16bf a = frag2(&Qs[m][kk + hi * 8], &Qs[m][kk + 16 + hi * 8]);
      v16bf bb = frag2(&Ks[m][kk + hi * 8], &Ks[m][kk + 16 + hi * 8]);
      c = wmma_bf16(a, bb, c);
    }
#pragma unroll
    for (int r = 0; r < 8; ++r) Ss[(lane < 16) ? r : (8 + r)][m] = c[r];
    __syncthreads();

    // online softmax (lane r owns query row r)
    if (lane < 16) {
      float mx = -3.0e38f;
      for (int j = 0; j < 16; ++j) mx = fmaxf(mx, Ss[lane][j]);
      float mnew = fmaxf(mrow[lane], mx);
      float rs = expf(mrow[lane] - mnew);
      float sum = 0.0f;
      for (int j = 0; j < 16; ++j) {
        float p = expf(Ss[lane][j] - mnew);
        Ps[lane][j] = (__bf16)p;
        sum += p;
      }
      lrow[lane] = lrow[lane] * rs + sum;
      mrow[lane] = mnew;
      for (int j = 0; j < DH_; ++j) Os[lane][j] *= rs;
    }
    __syncthreads();

    // O += P @ V  (A = P 16x32 zero-padded, B from transposed VsT rows)
    for (int c0 = 0; c0 < DH_; c0 += 16) {
      v8f acc;
#pragma unroll
      for (int r = 0; r < 8; ++r) acc[r] = Os[(lane < 16) ? r : (8 + r)][c0 + m];
      v16bf a = frag2(&Ps[m][hi * 8], &Ps[m][16 + hi * 8]);
      v16bf bb = frag2(&VsT[c0 + m][hi * 8], &VsT[c0 + m][16 + hi * 8]);
      acc = wmma_bf16(a, bb, acc);
#pragma unroll
      for (int r = 0; r < 8; ++r) Os[(lane < 16) ? r : (8 + r)][c0 + m] = acc[r];
    }
    __syncthreads();
  }

  for (int idx = lane; idx < 16 * DH_; idx += 32) {
    int r = idx / DH_, cc = idx - r * DH_;
    o[(size_t)(b * S + q0 + r) * NI_ + head * DH_ + cc] = Os[r][cc] / lrow[r];
  }
}

// ---------------------------------------------------------------------------
// LayerNorm, one wave per row (biased variance, eps=1e-5). float4 I/O.
// ---------------------------------------------------------------------------
__global__ __launch_bounds__(256) void layernorm_kernel(
    const float* __restrict__ in, const float* __restrict__ w,
    const float* __restrict__ bb, float* __restrict__ out, int N, int C) {
  const int lane = threadIdx.x & 31;
  const int wave = threadIdx.x >> 5;
  const int row = blockIdx.x * (blockDim.x >> 5) + wave;
  if (row >= N) return;
  const float* p = in + (size_t)row * C;
  float s = 0.0f, s2 = 0.0f;
  for (int c = lane * 4; c < C; c += 128) {
    float4 v = *(const float4*)&p[c];
    s += v.x + v.y + v.z + v.w;
    s2 += v.x * v.x + v.y * v.y + v.z * v.z + v.w * v.w;
  }
  for (int off = 16; off; off >>= 1) {
    s  += __shfl_xor(s,  off, 32);
    s2 += __shfl_xor(s2, off, 32);
  }
  const float mu = s / C;
  const float var = s2 / C - mu * mu;
  const float inv = rsqrtf(var + 1e-5f);
  for (int c = lane * 4; c < C; c += 128) {
    float4 v = *(const float4*)&p[c];
    float4 ww = *(const float4*)&w[c];
    float4 bv = *(const float4*)&bb[c];
    float4 r;
    r.x = (v.x - mu) * inv * ww.x + bv.x;
    r.y = (v.y - mu) * inv * ww.y + bv.y;
    r.z = (v.z - mu) * inv * ww.z + bv.z;
    r.w = (v.w - mu) * inv * ww.w + bv.w;
    *(float4*)&out[(size_t)row * C + c] = r;
  }
}

// ---------------------------------------------------------------------------
// Embedding lookup + positional encoding (reference's swapped pos/i roles:
// angle[s][c] = c * 10000^{-2s/C}; even seq row -> sin, odd -> cos).
// ---------------------------------------------------------------------------
__global__ void embed_kernel(const int* __restrict__ inp,
                             const float* __restrict__ emb,
                             float* __restrict__ x, int S, int C) {
  const int s = blockIdx.x, b = blockIdx.y;
  const int tok = inp[b * S + s];
  const float w = expf((-2.0f * s / C) * logf(10000.0f));
  for (int c = threadIdx.x; c < C; c += blockDim.x) {
    float ang = c * w;
    float pe = ((s & 1) == 0) ? sinf(ang) : cosf(ang);
    x[(size_t)(b * S + s) * C + c] = emb[(size_t)tok * C + c] + pe;
  }
}

// duration prefix sums (serial per batch; S=256, trivial)
__global__ void csum_kernel(const int* __restrict__ dur,
                            const float* __restrict__ ratio,
                            int* __restrict__ csum, int S) {
  const int b = blockIdx.x;
  if (threadIdx.x == 0) {
    int acc = 0;
    float r = ratio[0];
    for (int s = 0; s < S; ++s) {
      acc += (int)(r * (float)dur[b * S + s]);
      csum[b * S + s] = acc;
    }
  }
}

// length regulator gather (searchsorted right) + mel positional encoding
__global__ void gather_kernel(const float* __restrict__ xin,
                              const int* __restrict__ csum,
                              float* __restrict__ xout, int S, int T, int C) {
  const int t = blockIdx.x, b = blockIdx.y;
  __shared__ int tok_s;
  if (threadIdx.x == 0) {
    const int* cs = csum + b * S;
    int lo = 0, hi = S;
    while (lo < hi) {
      int mid = (lo + hi) >> 1;
      if (cs[mid] <= t) lo = mid + 1; else hi = mid;
    }
    tok_s = (lo < S - 1) ? lo : (S - 1);
  }
  __syncthreads();
  const int tok = tok_s;
  const float w = expf((-2.0f * t / C) * logf(10000.0f));
  for (int c = threadIdx.x; c < C; c += blockDim.x) {
    float ang = c * w;
    float pe = ((t & 1) == 0) ? sinf(ang) : cosf(ang);
    xout[(size_t)(b * T + t) * C + c] = xin[(size_t)(b * S + tok) * C + c] + pe;
  }
}

// ---------------------------------------------------------------------------
// Host orchestration
// ---------------------------------------------------------------------------
static inline int gemm_grid(int N, int M) {
  return (N >> 7) * ((M + 63) >> 6);
}

extern "C" void kernel_launch(void* const* d_in, const int* in_sizes, int n_in,
                              void* d_out, int out_size, void* d_ws, size_t ws_size,
                              hipStream_t stream) {
  (void)in_sizes; (void)n_in; (void)out_size; (void)ws_size;

  const int* inp     = (const int*)d_in[0];
  const int* dur     = (const int*)d_in[1];
  const float* ratio = (const float*)d_in[2];
  // d_in[3] = total_len (fixed: T)
  const float* emb   = (const float*)d_in[4];
  const float* lin_w = (const float*)d_in[25];
  const float* lin_b = (const float*)d_in[26];

  // workspace layout
  float* xA   = (float*)d_ws;                          // [4096, 384]
  float* xB   = xA  + (size_t)TT_ * NB_ * NI_;         // [4096, 384]
  float* hbuf = xB  + (size_t)TT_ * NB_ * NI_;         // [4096, 1152]
  float* obuf = hbuf + (size_t)TT_ * NB_ * NI3_;       // [4096, 384]
  float* tmp  = obuf + (size_t)TT_ * NB_ * NI_;        // [4096, 384]
  int*   csum = (int*)(tmp + (size_t)TT_ * NB_ * NI_); // [B*S]

  // ---- phoneme embedding + pos enc ----
  embed_kernel<<<dim3(SS_, NB_), 128, 0, stream>>>(inp, emb, xA, SS_, NI_);

  // ---- FFT stacks ----
  auto run_stack = [&](float* x, int Seq, int pbase) {
    const int N = NB_ * Seq;
    const float* kqv_w  = (const float*)d_in[pbase + 0];
    const float* kqv_b  = (const float*)d_in[pbase + 1];
    const float* proj_w = (const float*)d_in[pbase + 2];
    const float* proj_b = (const float*)d_in[pbase + 3];
    const float* n1w    = (const float*)d_in[pbase + 4];
    const float* n1b    = (const float*)d_in[pbase + 5];
    const float* conv_w = (const float*)d_in[pbase + 6];
    const float* conv_b = (const float*)d_in[pbase + 7];
    const float* n2w    = (const float*)d_in[pbase + 8];
    const float* n2b    = (const float*)d_in[pbase + 9];

    for (int l = 0; l < NLAY_; ++l) {
      const float* kw = kqv_w  + (size_t)l * NI_ * NI3_;
      const float* kb = kqv_b  + (size_t)l * NI3_;
      const float* pw = proj_w + (size_t)l * NI_ * NI_;
      const float* pb = proj_b + (size_t)l * NI_;
      const float* w1 = n1w + (size_t)l * NI_;
      const float* b1 = n1b + (size_t)l * NI_;
      const float* cw = conv_w + (size_t)l * NI_ * NI_ * 3;
      const float* cb = conv_b + (size_t)l * NI_;
      const float* w2 = n2w + (size_t)l * NI_;
      const float* b2 = n2b + (size_t)l * NI_;

      // h = x @ kqv_w + kqv_b
      gemm_wmma_kernel<<<gemm_grid(N, NI3_), 256, 0, stream>>>(
          x, N, NI_, kw, NI3_, kb, nullptr, hbuf, 0);
      // multi-head attention
      attn_wmma_kernel<<<dim3(Seq / 16, NH_, NB_), 32, 0, stream>>>(hbuf, obuf, Seq);
      // proj + residual, then LayerNorm
      gemm_wmma_kernel<<<gemm_grid(N, NI_), 256, 0, stream>>>(
          obuf, N, NI_, pw, NI_, pb, x, tmp, 0);
      layernorm_kernel<<<(N + 7) / 8, 256, 0, stream>>>(tmp, w1, b1, x, N, NI_);
      // conv (k=3, same-pad) + residual, then LayerNorm
      conv_wmma_kernel<<<gemm_grid(N, NI_), 256, 0, stream>>>(
          x, cw, cb, tmp, NB_, Seq, NI_);
      layernorm_kernel<<<(N + 7) / 8, 256, 0, stream>>>(tmp, w2, b2, x, N, NI_);
    }
  };

  run_stack(xA, SS_, 5);

  // ---- length regulator + mel pos enc ----
  csum_kernel<<<NB_, 32, 0, stream>>>(dur, ratio, csum, SS_);
  gather_kernel<<<dim3(TT_, NB_), 128, 0, stream>>>(xA, csum, xB, SS_, TT_, NI_);

  run_stack(xB, TT_, 15);

  // ---- final linear, stored transposed [B, NOUT, T] ----
  gemm_wmma_kernel<<<gemm_grid(NB_ * TT_, NOUT_), 256, 0, stream>>>(
      xB, NB_ * TT_, NI_, lin_w, NOUT_, lin_b, nullptr,
      (float*)d_out, TT_);
}
